// HSPool_46351287058765
// MI455X (gfx1250) — matrix operand order repaired and checked by reading, not compiled, not run on hardware
//
#include <hip/hip_runtime.h>

// ---------------- problem constants ----------------
constexpr int N_ = 20000;
constexpr int E_ = 320000;
constexpr int D_ = 128;
constexpr int G_ = 64;
constexpr int H_ = 4;
constexpr int S2S_STEPS = 64;
constexpr int ND_ = N_ * D_;

typedef __attribute__((ext_vector_type(16))) _Float16 v16h;
typedef __attribute__((ext_vector_type(8)))  float    v8f;

// ---------------- device helpers ----------------
static __device__ __forceinline__ float sigm_(float x) { return 1.0f / (1.0f + __expf(-x)); }
static __device__ __forceinline__ float silu_(float x) { return x * sigm_(x); }
// order-preserving float<->uint encoding for atomicMax on signed floats
static __device__ __forceinline__ unsigned fenc_(float f) {
  unsigned u = __float_as_uint(f);
  return (u & 0x80000000u) ? ~u : (u | 0x80000000u);
}
static __device__ __forceinline__ float fdec_(unsigned u) {
  unsigned b = (u & 0x80000000u) ? (u & 0x7FFFFFFFu) : ~u;
  return __uint_as_float(b);
}

// ---------------- generic zero ----------------
__global__ void k_zero(float* __restrict__ p, int n) {
  int i = blockIdx.x * blockDim.x + threadIdx.x;
  if (i < n) p[i] = 0.0f;
}

// ---------------- WMMA GEMM: C[N,128] = A[N,128] @ W[128,128] ----------------
// Block = 128 threads = 4 waves; each wave computes one 16-row x 128-col strip.
// W is converted f32->f16 and pre-swizzled into LDS in the exact B-fragment
// lane layout so each lane reads 16 contiguous halves per fragment.
__global__ __launch_bounds__(128) void k_gemm_nodes(
    const float* __restrict__ A, const float* __restrict__ W,
    float* __restrict__ Cout, int nrows) {
  __shared__ _Float16 bsw[4][8][32][16];  // [kChunk][nChunk][lane][t] = 32KB
  const int tid = threadIdx.x;
  for (int idx = tid; idx < 4 * 8 * 32 * 16; idx += 128) {
    int t    = idx & 15;
    int lane = (idx >> 4) & 31;
    int jc   = (idx >> 9) & 7;
    int kc   = idx >> 12;
    // B 32x16 f16 layout: lanes 0-15 -> col=lane, K=0..7,16..23; lanes 16-31 -> K=8..15,24..31
    int krow = kc * 32 + ((lane & 16) ? 8 : 0) + (t < 8 ? t : t + 8);
    int col  = jc * 16 + (lane & 15);
    bsw[kc][jc][lane][t] = (_Float16)W[krow * D_ + col];
  }
  __syncthreads();
  const int wave  = tid >> 5;
  const int lane  = tid & 31;
  const int strip = blockIdx.x * 4 + wave;
  if (strip * 16 >= nrows) return;  // N is a multiple of 16, guard whole strip
  const int row = strip * 16 + (lane & 15);
  const int hi  = lane >> 4;  // A layout: lanes 16-31 hold K offset +8
  v8f acc[8];
  for (int jc = 0; jc < 8; ++jc)
    for (int i = 0; i < 8; ++i) acc[jc][i] = 0.0f;
  const float* ap = A + (size_t)row * D_;
  for (int kc = 0; kc < 4; ++kc) {
    const int k0 = kc * 32 + hi * 8;
    float4 a0 = *(const float4*)(ap + k0);
    float4 a1 = *(const float4*)(ap + k0 + 4);
    float4 a2 = *(const float4*)(ap + k0 + 16);
    float4 a3 = *(const float4*)(ap + k0 + 20);
    v16h af;
    af[0]  = (_Float16)a0.x; af[1]  = (_Float16)a0.y; af[2]  = (_Float16)a0.z; af[3]  = (_Float16)a0.w;
    af[4]  = (_Float16)a1.x; af[5]  = (_Float16)a1.y; af[6]  = (_Float16)a1.z; af[7]  = (_Float16)a1.w;
    af[8]  = (_Float16)a2.x; af[9]  = (_Float16)a2.y; af[10] = (_Float16)a2.z; af[11] = (_Float16)a2.w;
    af[12] = (_Float16)a3.x; af[13] = (_Float16)a3.y; af[14] = (_Float16)a3.z; af[15] = (_Float16)a3.w;
    for (int jc = 0; jc < 8; ++jc) {
      v16h bf = *(const v16h*)(&bsw[kc][jc][lane][0]);
      acc[jc] = __builtin_amdgcn_wmma_f32_16x16x32_f16(
          false, af, false, bf, (short)0, acc[jc], false, false);
    }
  }
  // C layout: VGPR r -> M = r (lanes 0-15) / 8+r (lanes 16-31); N = lane&15
  const int rb = strip * 16 + hi * 8;
  const int cb = lane & 15;
  for (int jc = 0; jc < 8; ++jc)
    for (int r = 0; r < 8; ++r)
      Cout[(size_t)(rb + r) * D_ + jc * 16 + cb] = acc[jc][r];
}

// ---------------- LSTM cell (used for both Set2Set cells) ----------------
// thread per (g,d); gates columns {d, D+d, 2D+d, 3D+d}; optional fused resets.
__global__ void k_lstm(const float* __restrict__ inp, int kin,
                       const float* __restrict__ hin, float* __restrict__ hout,
                       float* __restrict__ cst,
                       const float* __restrict__ Wi, const float* __restrict__ Wh,
                       const float* __restrict__ bb,
                       float* r_rst, unsigned* m_rst, float* s_rst) {
  int idx = blockIdx.x * blockDim.x + threadIdx.x;
  if (idx >= G_ * D_) return;
  int g = idx / D_, d = idx - g * D_;
  const float* iv = inp + (size_t)g * kin;
  const float* hv = hin + (size_t)g * D_;
  float a0 = bb[d], a1 = bb[D_ + d], a2 = bb[2 * D_ + d], a3 = bb[3 * D_ + d];
  for (int k = 0; k < kin; ++k) {
    float xv = iv[k];
    const float* wr = Wi + (size_t)k * (4 * D_) + d;
    a0 += xv * wr[0]; a1 += xv * wr[D_]; a2 += xv * wr[2 * D_]; a3 += xv * wr[3 * D_];
  }
  for (int k = 0; k < D_; ++k) {
    float xv = hv[k];
    const float* wr = Wh + (size_t)k * (4 * D_) + d;
    a0 += xv * wr[0]; a1 += xv * wr[D_]; a2 += xv * wr[2 * D_]; a3 += xv * wr[3 * D_];
  }
  float ig = sigm_(a0), fg = sigm_(a1), gg = tanhf(a2), og = sigm_(a3);
  float cn = fg * cst[idx] + ig * gg;
  cst[idx] = cn;
  hout[idx] = og * tanhf(cn);
  if (r_rst) {  // reset per-step attention scratch (runs before it is used)
    r_rst[idx] = 0.0f;
    if (d == 0) { m_rst[g] = 0u; s_rst[g] = 0.0f; }
  }
}

// e[n] = dot(x[n], q[batch[n]]); segment max via ordered-uint atomicMax
__global__ void k_s2s_e(const float* __restrict__ x, const int* __restrict__ batch,
                        const float* __restrict__ q, float* __restrict__ ebuf,
                        unsigned* __restrict__ gmax) {
  int n = blockIdx.x * blockDim.x + threadIdx.x;
  if (n >= N_) return;
  int g = batch[n];
  const float* xr = x + (size_t)n * D_;
  const float* qr = q + (size_t)g * D_;
  float e = 0.0f;
  for (int k = 0; k < D_; ++k) e += xr[k] * qr[k];
  ebuf[n] = e;
  atomicMax(gmax + g, fenc_(e));
}

// v[i] = exp(v[i] - max[seg]); atomic sum per segment (shared by s2s & readout)
__global__ void k_seg_exp(float* __restrict__ v, const int* __restrict__ seg,
                          const unsigned* __restrict__ mx, float* __restrict__ sm, int n) {
  int i = blockIdx.x * blockDim.x + threadIdx.x;
  if (i >= n) return;
  int g = seg[i];
  float a = __expf(v[i] - fdec_(mx[g]));
  v[i] = a;
  atomicAdd(sm + g, a);
}

// r[g][:] += a[n] * src[n][:]  (thread = (node, quarter of D))
__global__ void k_seg_acc(const float* __restrict__ src, const int* __restrict__ batch,
                          const float* __restrict__ a, float* __restrict__ r) {
  int idx = blockIdx.x * blockDim.x + threadIdx.x;
  if (idx >= N_ * 4) return;
  int n = idx >> 2, p = idx & 3;
  int g = batch[n];
  float av = a[n];
  const float* sr = src + (size_t)n * D_ + p * 32;
  float* rr = r + (size_t)g * D_ + p * 32;
  for (int c = 0; c < 32; ++c) atomicAdd(rr + c, av * sr[c]);
}

__global__ void k_s2s_qstar(const float* __restrict__ q, const float* __restrict__ r,
                            const float* __restrict__ gsum, float* __restrict__ qstar) {
  int idx = blockIdx.x * blockDim.x + threadIdx.x;
  if (idx >= G_ * D_) return;
  int g = idx / D_, d = idx - g * D_;
  qstar[(size_t)g * 2 * D_ + d] = q[idx];
  qstar[(size_t)g * 2 * D_ + D_ + d] = r[idx] / (gsum[g] + 1e-16f);
}

// ---------------- GAT ----------------
// per (n,h): attention coefficients + reset of softmax scratch and agg output
__global__ void k_attn_coef(const float* __restrict__ hW,
                            const float* __restrict__ a_src, const float* __restrict__ a_dst,
                            float* __restrict__ as, float* __restrict__ ad,
                            unsigned* __restrict__ nmax, float* __restrict__ nsum,
                            float* __restrict__ agg) {
  int idx = blockIdx.x * blockDim.x + threadIdx.x;
  if (idx >= N_ * H_) return;
  int n = idx >> 2, h = idx & 3;
  const float* hw = hW + (size_t)n * D_ + h * 32;
  const float* ws = a_src + h * 32;
  const float* wd = a_dst + h * 32;
  float s = 0.0f, t = 0.0f;
  for (int c = 0; c < 32; ++c) { s += hw[c] * ws[c]; t += hw[c] * wd[c]; }
  as[idx] = s; ad[idx] = t;
  nmax[idx] = 0u; nsum[idx] = 0.0f;
  float* ag = agg + (size_t)n * D_ + h * 32;
  for (int c = 0; c < 32; ++c) ag[c] = 0.0f;
}

__global__ void k_edge_max(const int* __restrict__ eidx,
                           const float* __restrict__ as, const float* __restrict__ ad,
                           float* __restrict__ alpha, unsigned* __restrict__ nmax) {
  int idx = blockIdx.x * blockDim.x + threadIdx.x;
  if (idx >= E_ * H_) return;
  int e = idx >> 2, h = idx & 3;
  int s = eidx[e], t = eidx[E_ + e];
  float v = as[s * H_ + h] + ad[t * H_ + h];
  v = (v > 0.0f) ? v : 0.2f * v;  // leaky_relu
  alpha[idx] = v;
  atomicMax(nmax + t * H_ + h, fenc_(v));
}

__global__ void k_edge_exp(const int* __restrict__ eidx, float* __restrict__ alpha,
                           const unsigned* __restrict__ nmax, float* __restrict__ nsum) {
  int idx = blockIdx.x * blockDim.x + threadIdx.x;
  if (idx >= E_ * H_) return;
  int e = idx >> 2, h = idx & 3;
  int t = eidx[E_ + e];
  float a = __expf(alpha[idx] - fdec_(nmax[t * H_ + h]));
  alpha[idx] = a;
  atomicAdd(nsum + t * H_ + h, a);
}

__global__ void k_edge_agg(const int* __restrict__ eidx, const float* __restrict__ alpha,
                           const float* __restrict__ nsum, const float* __restrict__ hW,
                           float* __restrict__ agg) {
  int idx = blockIdx.x * blockDim.x + threadIdx.x;
  if (idx >= E_ * H_) return;
  int e = idx >> 2, h = idx & 3;
  int s = eidx[e], t = eidx[E_ + e];
  float w = alpha[idx] / (nsum[t * H_ + h] + 1e-16f);
  const float* hs = hW + (size_t)s * D_ + h * 32;
  float* ad = agg + (size_t)t * D_ + h * 32;
  for (int c = 0; c < 32; ++c) atomicAdd(ad + c, w * hs[c]);
}

__global__ void k_bias(float* __restrict__ h, const float* __restrict__ bias) {
  int idx = blockIdx.x * blockDim.x + threadIdx.x;
  if (idx >= ND_) return;
  h[idx] += bias[idx & (D_ - 1)];
}

// ---------------- gated readout + per-layer MLP ----------------
__global__ void k_pool_reset(float* __restrict__ pool, unsigned* __restrict__ gmax,
                             float* __restrict__ gsum) {
  int idx = blockIdx.x * blockDim.x + threadIdx.x;
  if (idx >= G_ * D_) return;
  pool[idx] = 0.0f;
  if ((idx & (D_ - 1)) == 0) { gmax[idx / D_] = 0u; gsum[idx / D_] = 0.0f; }
}

__global__ void k_score(const float* __restrict__ h, const float* __restrict__ gW,
                        const float* __restrict__ gb, const int* __restrict__ batch,
                        float* __restrict__ sc, unsigned* __restrict__ gmax) {
  int n = blockIdx.x * blockDim.x + threadIdx.x;
  if (n >= N_) return;
  const float* hr = h + (size_t)n * D_;
  float s = gb[0];
  for (int k = 0; k < D_; ++k) s += hr[k] * gW[k];
  sc[n] = s;
  atomicMax(gmax + batch[n], fenc_(s));
}

// t1 = silu(concat(gx, pool/gsum) @ fW1 + fb1)
__global__ void k_loc1(const float* __restrict__ qstar, const float* __restrict__ pool,
                       const float* __restrict__ gsum, const float* __restrict__ fW1,
                       const float* __restrict__ fb1, float* __restrict__ t1) {
  int idx = blockIdx.x * blockDim.x + threadIdx.x;
  if (idx >= G_ * D_) return;
  int g = idx / D_, d = idx - g * D_;
  float acc = fb1[d];
  const float* gx = qstar + (size_t)g * 2 * D_;
  for (int k = 0; k < 2 * D_; ++k) acc += gx[k] * fW1[(size_t)k * D_ + d];
  float inv = 1.0f / (gsum[g] + 1e-16f);
  const float* pr = pool + (size_t)g * D_;
  for (int k = 0; k < D_; ++k) acc += pr[k] * inv * fW1[(size_t)(2 * D_ + k) * D_ + d];
  t1[idx] = silu_(acc);
}

// cat5[:, lofs:lofs+D] = t1 @ fW2 + fb2
__global__ void k_loc2(const float* __restrict__ t1, const float* __restrict__ fW2,
                       const float* __restrict__ fb2, float* __restrict__ cat5, int lofs) {
  int idx = blockIdx.x * blockDim.x + threadIdx.x;
  if (idx >= G_ * D_) return;
  int g = idx / D_, d = idx - g * D_;
  float acc = fb2[d];
  const float* tr = t1 + (size_t)g * D_;
  for (int k = 0; k < D_; ++k) acc += tr[k] * fW2[(size_t)k * D_ + d];
  cat5[(size_t)g * 5 * D_ + lofs + d] = acc;
}

__global__ void k_cat_gx(const float* __restrict__ qstar, float* __restrict__ cat5) {
  int idx = blockIdx.x * blockDim.x + threadIdx.x;
  if (idx >= G_ * 2 * D_) return;
  int g = idx / (2 * D_), j = idx - g * 2 * D_;
  cat5[(size_t)g * 5 * D_ + 3 * D_ + j] = qstar[idx];
}

__global__ void k_fin1(const float* __restrict__ cat5, const float* __restrict__ oW1,
                       const float* __restrict__ ob1, float* __restrict__ t2) {
  int idx = blockIdx.x * blockDim.x + threadIdx.x;
  if (idx >= G_ * 4 * D_) return;
  int g = idx >> 9, d = idx & 511;
  float acc = ob1[d];
  const float* cr = cat5 + (size_t)g * 5 * D_;
  for (int k = 0; k < 5 * D_; ++k) acc += cr[k] * oW1[(size_t)k * (4 * D_) + d];
  t2[idx] = silu_(acc);
}

__global__ void k_fin2(const float* __restrict__ t2, const float* __restrict__ oW2,
                       const float* __restrict__ ob2, float* __restrict__ out) {
  int idx = blockIdx.x * blockDim.x + threadIdx.x;
  if (idx >= G_ * 3 * D_) return;
  int g = idx / (3 * D_), d = idx - g * 3 * D_;
  float acc = ob2[d];
  const float* tr = t2 + (size_t)g * 4 * D_;
  for (int k = 0; k < 4 * D_; ++k) acc += tr[k] * oW2[(size_t)k * (3 * D_) + d];
  out[idx] = acc;
}

// ---------------- host launcher ----------------
extern "C" void kernel_launch(void* const* d_in, const int* in_sizes, int n_in,
                              void* d_out, int out_size, void* d_ws, size_t ws_size,
                              hipStream_t stream) {
  (void)in_sizes; (void)n_in; (void)out_size; (void)ws_size;
  // Inputs in setup_inputs() insertion order (pytree leaves in dict order).
  const float* x     = (const float*)d_in[0];
  const int*   eidx  = (const int*)d_in[1];
  const int*   batch = (const int*)d_in[2];
  const float* Wi1 = (const float*)d_in[3];
  const float* Wh1 = (const float*)d_in[4];
  const float* b1  = (const float*)d_in[5];
  const float* Wi2 = (const float*)d_in[6];
  const float* Wh2 = (const float*)d_in[7];
  const float* b2  = (const float*)d_in[8];
  const float *LW[3], *Lasrc[3], *Ladst[3], *Lbias[3], *LgW[3], *Lgb[3],
              *LfW1[3], *Lfb1[3], *LfW2[3], *Lfb2[3];
  for (int l = 0; l < 3; ++l) {
    int b = 9 + l * 10;
    LW[l]    = (const float*)d_in[b + 0];
    Lasrc[l] = (const float*)d_in[b + 1];
    Ladst[l] = (const float*)d_in[b + 2];
    Lbias[l] = (const float*)d_in[b + 3];
    LgW[l]   = (const float*)d_in[b + 4];
    Lgb[l]   = (const float*)d_in[b + 5];
    LfW1[l]  = (const float*)d_in[b + 6];
    Lfb1[l]  = (const float*)d_in[b + 7];
    LfW2[l]  = (const float*)d_in[b + 8];
    Lfb2[l]  = (const float*)d_in[b + 9];
  }
  const float* oW1 = (const float*)d_in[39];
  const float* ob1 = (const float*)d_in[40];
  const float* oW2 = (const float*)d_in[41];
  const float* ob2 = (const float*)d_in[42];

  // workspace carve-out (f32 units)
  float* ws = (float*)d_ws;
  size_t o = 0;
  auto take = [&](size_t n) { float* p = ws + o; o += n; return p; };
  float* hbuf0 = take(ND_);
  float* hbuf1 = take(ND_);
  float* hW    = take(ND_);
  float* a_s   = take((size_t)N_ * H_);
  float* a_d   = take((size_t)N_ * H_);
  unsigned* nmaxu = (unsigned*)take((size_t)N_ * H_);
  float* nsum  = take((size_t)N_ * H_);
  float* alpha = take((size_t)E_ * H_);
  float* qstar = take((size_t)G_ * 2 * D_);  // contiguous with h1a..c2 (zeroed once)
  float* h1a = take((size_t)G_ * D_);
  float* h1b = take((size_t)G_ * D_);
  float* c1  = take((size_t)G_ * D_);
  float* h2a = take((size_t)G_ * D_);
  float* h2b = take((size_t)G_ * D_);
  float* c2  = take((size_t)G_ * D_);
  unsigned* gmaxu = (unsigned*)take(G_);
  float* gsum = take(G_);
  float* rbuf = take((size_t)G_ * D_);
  float* ebuf = take(N_);
  float* sc   = take(N_);
  float* pool = take((size_t)G_ * D_);
  float* t1   = take((size_t)G_ * D_);
  float* cat5 = take((size_t)G_ * 5 * D_);
  float* t2   = take((size_t)G_ * 4 * D_);
  float* out  = (float*)d_out;

  auto cdiv = [](int a, int b) { return (a + b - 1) / b; };
  const int TB = 256;

  // ---- Set2Set: zero LSTM state (qstar..c2 are contiguous: 8*G*D floats) ----
  k_zero<<<cdiv(G_ * 8 * D_, TB), TB, 0, stream>>>(qstar, G_ * 8 * D_);
  float *h1in = h1a, *h1out = h1b, *h2in = h2a, *h2out = h2b;
  for (int s = 0; s < S2S_STEPS; ++s) {
    k_lstm<<<cdiv(G_ * D_, TB), TB, 0, stream>>>(qstar, 2 * D_, h1in, h1out, c1,
                                                 Wi1, Wh1, b1, nullptr, nullptr, nullptr);
    k_lstm<<<cdiv(G_ * D_, TB), TB, 0, stream>>>(h1out, D_, h2in, h2out, c2,
                                                 Wi2, Wh2, b2, rbuf, gmaxu, gsum);
    k_s2s_e<<<cdiv(N_, TB), TB, 0, stream>>>(x, batch, h2out, ebuf, gmaxu);
    k_seg_exp<<<cdiv(N_, TB), TB, 0, stream>>>(ebuf, batch, gmaxu, gsum, N_);
    k_seg_acc<<<cdiv(N_ * 4, TB), TB, 0, stream>>>(x, batch, ebuf, rbuf);
    k_s2s_qstar<<<cdiv(G_ * D_, TB), TB, 0, stream>>>(h2out, rbuf, gsum, qstar);
    float* t;
    t = h1in; h1in = h1out; h1out = t;
    t = h2in; h2in = h2out; h2out = t;
  }
  // gx == qstar now

  // ---- 3 GAT layers + readout ----
  const float* hin = x;
  float* hout = hbuf0;
  for (int l = 0; l < 3; ++l) {
    k_gemm_nodes<<<cdiv(N_ / 16, 4), 128, 0, stream>>>(hin, LW[l], hW, N_);
    k_attn_coef<<<cdiv(N_ * H_, TB), TB, 0, stream>>>(hW, Lasrc[l], Ladst[l],
                                                      a_s, a_d, nmaxu, nsum, hout);
    k_edge_max<<<cdiv(E_ * H_, TB), TB, 0, stream>>>(eidx, a_s, a_d, alpha, nmaxu);
    k_edge_exp<<<cdiv(E_ * H_, TB), TB, 0, stream>>>(eidx, alpha, nmaxu, nsum);
    k_edge_agg<<<cdiv(E_ * H_, TB), TB, 0, stream>>>(eidx, alpha, nsum, hW, hout);
    k_bias<<<cdiv(ND_, TB), TB, 0, stream>>>(hout, Lbias[l]);
    // gated attention pooling
    k_pool_reset<<<cdiv(G_ * D_, TB), TB, 0, stream>>>(pool, gmaxu, gsum);
    k_score<<<cdiv(N_, TB), TB, 0, stream>>>(hout, LgW[l], Lgb[l], batch, sc, gmaxu);
    k_seg_exp<<<cdiv(N_, TB), TB, 0, stream>>>(sc, batch, gmaxu, gsum, N_);
    k_seg_acc<<<cdiv(N_ * 4, TB), TB, 0, stream>>>(hout, batch, sc, pool);
    // per-layer MLP into cat5 columns [l*D, (l+1)*D)
    k_loc1<<<cdiv(G_ * D_, TB), TB, 0, stream>>>(qstar, pool, gsum, LfW1[l], Lfb1[l], t1);
    k_loc2<<<cdiv(G_ * D_, TB), TB, 0, stream>>>(t1, LfW2[l], Lfb2[l], cat5, l * D_);
    hin = hout;
    hout = (l == 0) ? hbuf1 : hbuf0;
  }

  // ---- final MLP ----
  k_cat_gx<<<cdiv(G_ * 2 * D_, TB), TB, 0, stream>>>(qstar, cat5);
  k_fin1<<<cdiv(G_ * 4 * D_, TB), TB, 0, stream>>>(cat5, oW1, ob1, t2);
  k_fin2<<<cdiv(G_ * 3 * D_, TB), TB, 0, stream>>>(t2, oW2, ob2, out);
}
